// GATConv_12025908428986
// MI455X (gfx1250) — matrix operand order repaired and checked by reading, not compiled, not run on hardware
//
#include <hip/hip_runtime.h>
#include <hip/hip_bf16.h>

#define N_NODES   10000
#define N_EDGES   320000
#define IN_FEATS  256
#define NUM_HEADS 8
#define OUT_FEATS 64
#define HD        (NUM_HEADS * OUT_FEATS)   // 512
#define NEG_SLOPE 0.2f

typedef __attribute__((ext_vector_type(16))) _Float16 v16h;
typedef __attribute__((ext_vector_type(8)))  _Float16 v8h;
typedef __attribute__((ext_vector_type(8)))  float    v8f;

// ---------------- workspace layout (bytes) ----------------
#define OFF_FT     0ull                       // 10000*512 f32   = 20,480,000
#define OFF_EL     20480000ull                // 10000*8 f32     =    320,000
#define OFF_ER     20800000ull
#define OFF_M      21120000ull
#define OFF_S      21440000ull
#define OFF_EW     21760000ull                // 320000*8 f32    = 10,240,000
#define OFF_FEATH  32000000ull                // 10000*256 f16   =  5,120,000
#define OFF_WTH    37120000ull                // 512*256 f16     =    262,144

// ---------------- helpers ----------------
__device__ __forceinline__ void atomicMaxF(float* addr, float v) {
    // classic monotone-bits trick, correct for mixed signs
    if (v >= 0.0f) atomicMax((int*)addr, __float_as_int(v));
    else           atomicMin((unsigned int*)addr, __float_as_uint(v));
}

// ---------------- kernel 1: f32 -> f16 convert (+ transpose W) ----------------
__global__ __launch_bounds__(256)
void k_convert(const float* __restrict__ feat, const float* __restrict__ W,
               _Float16* __restrict__ featH, _Float16* __restrict__ WtH) {
    int tid = blockIdx.x * blockDim.x + threadIdx.x;
    if (tid < N_NODES * IN_FEATS)
        featH[tid] = (_Float16)feat[tid];
    if (tid < IN_FEATS * HD) {            // 131072 elements of W [256][512]
        int k = tid >> 9;                 // row in W   (0..255)
        int c = tid & 511;                // col in W   (0..511)
        WtH[c * IN_FEATS + k] = (_Float16)W[tid];   // Wt [512][256]
    }
}

// ---------------- kernel 2: init out / m / s ----------------
__global__ __launch_bounds__(256)
void k_init(float* __restrict__ out, float* __restrict__ m, float* __restrict__ s) {
    int tid = blockIdx.x * blockDim.x + threadIdx.x;
    if (tid < N_NODES * HD) out[tid] = 0.0f;
    if (tid < N_NODES * NUM_HEADS) { m[tid] = -1.0e30f; s[tid] = 0.0f; }
}

// ---------------- kernel 3: WMMA GEMM ft = feat @ W, fused el/er ----------------
// grid = 625 blocks (one 16-row M-tile each), 256 threads = 8 waves,
// wave w computes the 16x64 tile of head h = w.
__global__ __launch_bounds__(256)
void k_gemm(const _Float16* __restrict__ featH, const _Float16* __restrict__ WtH,
            const float* __restrict__ attn_l, const float* __restrict__ attn_r,
            float* __restrict__ ft, float* __restrict__ el, float* __restrict__ er) {
    const int wave  = threadIdx.x >> 5;
    const int lane  = threadIdx.x & 31;
    const int l15   = lane & 15;
    const int lh    = lane >> 4;                // 0 or 1 (half-wave)
    const int mbase = blockIdx.x * 16;
    const int h     = wave;                     // head index
    const int nbase = h * OUT_FEATS;            // column base in [0,512)

    v8f acc[4] = {v8f{}, v8f{}, v8f{}, v8f{}};
    const int am = mbase + l15;                 // A row this lane supplies

    #pragma unroll
    for (int kk = 0; kk < IN_FEATS / 32; ++kk) {
        // A fragment: 16x32 f16. lanes 0-15: M=lane, K in {b..b+7, b+16..b+23}, b=8*lh
        union { v16h v; v8h h8[2]; } A;
        const _Float16* ap = featH + am * IN_FEATS + kk * 32 + lh * 8;
        A.h8[0] = *(const v8h*)(ap);
        A.h8[1] = *(const v8h*)(ap + 16);
        #pragma unroll
        for (int t = 0; t < 4; ++t) {
            // B fragment: 32x16 f16. lane: N=nbase+t*16+l15, K in {b..b+7, b+16..b+23}, b=8*lh
            union { v16h v; v8h h8[2]; } B;
            const _Float16* bp = WtH + (nbase + t * 16 + l15) * IN_FEATS + kk * 32 + lh * 8;
            B.h8[0] = *(const v8h*)(bp);
            B.h8[1] = *(const v8h*)(bp + 16);
            acc[t] = __builtin_amdgcn_wmma_f32_16x16x32_f16(
                false, A.v, false, B.v, (short)0, acc[t], false, false);
        }
    }

    // epilogue: store ft, fused el/er = sum_d ft * attn
    float pl[8], pr[8];
    #pragma unroll
    for (int r = 0; r < 8; ++r) { pl[r] = 0.0f; pr[r] = 0.0f; }
    #pragma unroll
    for (int t = 0; t < 4; ++t) {
        const int col = nbase + t * 16 + l15;
        const float al = attn_l[h * OUT_FEATS + t * 16 + l15];
        const float ar = attn_r[h * OUT_FEATS + t * 16 + l15];
        #pragma unroll
        for (int r = 0; r < 8; ++r) {
            const int mrow = mbase + r + lh * 8;      // C/D layout: M = r + 8*(lane>=16)
            const float v = acc[t][r];
            ft[mrow * HD + col] = v;
            pl[r] += v * al;
            pr[r] += v * ar;
        }
    }
    // reduce the 4-column partials across the 16 lanes of each half-wave
    #pragma unroll
    for (int off = 8; off >= 1; off >>= 1) {
        #pragma unroll
        for (int r = 0; r < 8; ++r) {
            pl[r] += __shfl_xor(pl[r], off, 16);
            pr[r] += __shfl_xor(pr[r], off, 16);
        }
    }
    if (l15 == 0) {
        #pragma unroll
        for (int r = 0; r < 8; ++r) {
            const int mrow = mbase + r + lh * 8;
            el[mrow * NUM_HEADS + h] = pl[r];
            er[mrow * NUM_HEADS + h] = pr[r];
        }
    }
}

// ---------------- kernel 4: edge logits + LeakyReLU + segment max ----------------
__global__ __launch_bounds__(256)
void k_edge_logits(const float* __restrict__ el, const float* __restrict__ er,
                   const int* __restrict__ src, const int* __restrict__ dst,
                   float* __restrict__ ew, float* __restrict__ m) {
    int tid = blockIdx.x * blockDim.x + threadIdx.x;
    if (tid >= N_EDGES * NUM_HEADS) return;
    int e = tid >> 3, h = tid & 7;
    int si = src[e], di = dst[e];
    float v = el[si * NUM_HEADS + h] + er[di * NUM_HEADS + h];
    v = v > 0.0f ? v : NEG_SLOPE * v;
    ew[tid] = v;
    atomicMaxF(&m[di * NUM_HEADS + h], v);
}

// ---------------- kernel 5: exp(e - m[dst]) + segment sum ----------------
__global__ __launch_bounds__(256)
void k_edge_exp(const int* __restrict__ dst, const float* __restrict__ m,
                float* __restrict__ ew, float* __restrict__ s) {
    int tid = blockIdx.x * blockDim.x + threadIdx.x;
    if (tid >= N_EDGES * NUM_HEADS) return;
    int e = tid >> 3, h = tid & 7;
    int di = dst[e];
    float ex = __expf(ew[tid] - m[di * NUM_HEADS + h]);
    ew[tid] = ex;                                  // overwrite in place
    atomicAdd(&s[di * NUM_HEADS + h], ex);
}

// ---------------- kernel 6: normalize + adjacency mask + scatter aggregate ------
// one wave per (edge, head): lanes cover the 64 feature dims (2 each).
__global__ __launch_bounds__(256)
void k_aggregate(const float* __restrict__ ft, const float* __restrict__ ew,
                 const float* __restrict__ s, const float* __restrict__ adj,
                 const int* __restrict__ src, const int* __restrict__ dst,
                 const int* __restrict__ idxp, float* __restrict__ out) {
    long long gw = (long long)blockIdx.x * (blockDim.x >> 5) + (threadIdx.x >> 5);
    int lane = threadIdx.x & 31;
    if (gw >= (long long)N_EDGES * NUM_HEADS) return;
    int e = (int)(gw >> 3), h = (int)(gw & 7);
    int si = src[e], di = dst[e];
    float coef = 0.0f;
    if (lane == 0) {
        int idx = *idxp;
        float ex = ew[e * NUM_HEADS + h];
        float ss = s[di * NUM_HEADS + h];
        float ad = adj[(long long)(si + idx) * N_NODES + (long long)(di + idx)];
        coef = (ex / ss) * ad;
    }
    coef = __shfl(coef, 0, 32);
    const float* f = ft + (long long)si * HD + h * OUT_FEATS;
    float*       o = out + (long long)di * HD + h * OUT_FEATS;
    atomicAdd(&o[lane],      f[lane]      * coef);
    atomicAdd(&o[lane + 32], f[lane + 32] * coef);
}

// ---------------- launcher ----------------
extern "C" void kernel_launch(void* const* d_in, const int* in_sizes, int n_in,
                              void* d_out, int out_size, void* d_ws, size_t ws_size,
                              hipStream_t stream) {
    (void)in_sizes; (void)n_in; (void)out_size; (void)ws_size;
    const float* feat   = (const float*)d_in[0];
    const float* W      = (const float*)d_in[1];
    const float* attn_l = (const float*)d_in[2];
    const float* attn_r = (const float*)d_in[3];
    const float* adj    = (const float*)d_in[4];
    const int*   src    = (const int*)d_in[5];
    const int*   dst    = (const int*)d_in[6];
    const int*   idxp   = (const int*)d_in[7];
    float* out = (float*)d_out;

    char* ws = (char*)d_ws;
    float*    ft    = (float*)(ws + OFF_FT);
    float*    el    = (float*)(ws + OFF_EL);
    float*    er    = (float*)(ws + OFF_ER);
    float*    m     = (float*)(ws + OFF_M);
    float*    s     = (float*)(ws + OFF_S);
    float*    ew    = (float*)(ws + OFF_EW);
    _Float16* featH = (_Float16*)(ws + OFF_FEATH);
    _Float16* WtH   = (_Float16*)(ws + OFF_WTH);

    // 1) convert feat and W(transposed) to f16
    {
        int n = N_NODES * IN_FEATS;                 // 2,560,000 (covers W's 131,072 too)
        k_convert<<<(n + 255) / 256, 256, 0, stream>>>(feat, W, featH, WtH);
    }
    // 2) init out / m / s
    {
        int n = N_NODES * HD;                       // 5,120,000
        k_init<<<(n + 255) / 256, 256, 0, stream>>>(out, m, s);
    }
    // 3) WMMA GEMM + fused attention logits
    k_gemm<<<N_NODES / 16, 256, 0, stream>>>(featH, WtH, attn_l, attn_r, ft, el, er);
    // 4) edge logits + segment max
    {
        int n = N_EDGES * NUM_HEADS;                // 2,560,000
        k_edge_logits<<<(n + 255) / 256, 256, 0, stream>>>(el, er, src, dst, ew, m);
    }
    // 5) exp + segment sum
    {
        int n = N_EDGES * NUM_HEADS;
        k_edge_exp<<<(n + 255) / 256, 256, 0, stream>>>(dst, m, ew, s);
    }
    // 6) aggregate: one wave per (edge, head)
    {
        long long waves = (long long)N_EDGES * NUM_HEADS;   // 2,560,000 waves
        int wavesPerBlock = 256 / 32;
        long long blocks = (waves + wavesPerBlock - 1) / wavesPerBlock;  // 320,000
        k_aggregate<<<(unsigned)blocks, 256, 0, stream>>>(ft, ew, s, adj, src, dst, idxp, out);
    }
}
